// SEQLABEL_66967130079321
// MI455X (gfx1250) — matrix operand order repaired and checked by reading, not compiled, not run on hardware
//
#include <hip/hip_runtime.h>
#include <math.h>

// ---- problem constants (match reference) ----
#define B_    32
#define T_    512
#define V_    50000
#define E_    256
#define H_    512
#define L_    50
#define KIN_  768      // E + H
#define N4H_  2048     // 4*H
#define LPAD_ 64       // L padded to 4 WMMA N-tiles
#define NWG_  64       // persistent LSTM workgroups
#define UPW_  8        // hidden units per workgroup (NWG_*UPW_ == H_)
#define CPW_  32       // gate columns per workgroup (4*UPW_)
#define FORGET_BIAS 1.0f

#if __has_builtin(__builtin_amdgcn_global_load_async_to_lds_b128)
#define HAVE_ASYNC_LDS 1
#else
#define HAVE_ASYNC_LDS 0
#endif

typedef __bf16 bf16t;
typedef bf16t v16bf __attribute__((ext_vector_type(16)));
typedef float v8f   __attribute__((ext_vector_type(8)));

// types for the async global->LDS builtin (param type per hipcc diagnostic:
// "int __vector(4) __device__ *")
typedef int v4i_ __attribute__((vector_size(16)));
typedef __attribute__((address_space(1))) v4i_* gv4p;
typedef __attribute__((address_space(3))) v4i_* lv4p;

union Frag { uint4 u[2]; v16bf v; };

__device__ __forceinline__ unsigned short f2bf(float f) {
  unsigned u = __float_as_uint(f);
  u += 0x7fffu + ((u >> 16) & 1u);   // round-to-nearest-even
  return (unsigned short)(u >> 16);
}
__device__ __forceinline__ float sigm(float x) { return 1.0f / (1.0f + __expf(-x)); }

__device__ __forceinline__ v8f wmma_bf16(const Frag& a, const Frag& b, v8f c) {
  // (neg_a, A, neg_b, B, c_mod, C, reuse_a, reuse_b)
  return __builtin_amdgcn_wmma_f32_16x16x32_bf16(false, a.v, false, b.v, (short)0, c, false, false);
}

// async global -> LDS copy of one 16B chunk per lane (falls back to plain copy)
__device__ __forceinline__ void copy16_g2l(unsigned short* lds_dst,
                                           const unsigned short* gsrc) {
#if HAVE_ASYNC_LDS
  __builtin_amdgcn_global_load_async_to_lds_b128(
      (gv4p)(uintptr_t)gsrc,
      (lv4p)(unsigned)(uintptr_t)lds_dst,
      0, 0);
#else
  *(uint4*)lds_dst = *(const uint4*)gsrc;
#endif
}

__device__ __forceinline__ void wait_async_copies() {
#if HAVE_ASYNC_LDS
#if __has_builtin(__builtin_amdgcn_s_wait_asynccnt)
  __builtin_amdgcn_s_wait_asynccnt(0);
#else
  asm volatile("s_wait_asynccnt 0x0" ::: "memory");
#endif
#endif
}

// =====================================================================
// Kernel 0: weight conversion / transpose + state init
//   Wt  [N4H_][KIN_] bf16 : Wt[c*768+k] = lstm_W[k*2048+c]
//   Pt  [LPAD_][H_]  bf16 : Pt[c*512+k] = (c<50) ? proj_W[k*50+c] : 0
//   Hb  [2][B_][H_]  bf16 : zeros (double-buffered recurrent state)
//   ctr                    : grid-barrier counter = 0
// =====================================================================
__global__ void seqlab_init_kernel(const float* __restrict__ lstm_W,
                                   const float* __restrict__ proj_W,
                                   unsigned short* __restrict__ Wt,
                                   unsigned short* __restrict__ Pt,
                                   unsigned short* __restrict__ Hb,
                                   unsigned*       __restrict__ ctr) {
  size_t idx = (size_t)blockIdx.x * blockDim.x + threadIdx.x;
  if (idx < (size_t)N4H_ * KIN_) {
    int c = (int)(idx / KIN_), k = (int)(idx % KIN_);
    Wt[idx] = f2bf(lstm_W[(size_t)k * N4H_ + c]);
  }
  if (idx < (size_t)LPAD_ * H_) {
    int c = (int)(idx / H_), k = (int)(idx % H_);
    Pt[idx] = (c < L_) ? f2bf(proj_W[(size_t)k * L_ + c]) : (unsigned short)0;
  }
  if (idx < (size_t)2 * B_ * H_) Hb[idx] = 0;
  if (idx == 0) *ctr = 0u;
}

// =====================================================================
// Grid-wide barrier (persistent kernel; all NWG_ groups resident)
// =====================================================================
__device__ __forceinline__ void grid_barrier(unsigned* ctr, unsigned target) {
  __threadfence();
  __syncthreads();
  if (threadIdx.x == 0) {
    __hip_atomic_fetch_add(ctr, 1u, __ATOMIC_RELEASE, __HIP_MEMORY_SCOPE_AGENT);
    while (__hip_atomic_load(ctr, __ATOMIC_ACQUIRE, __HIP_MEMORY_SCOPE_AGENT) < target) {
      __builtin_amdgcn_s_sleep(2);
    }
  }
  __syncthreads();
}

// =====================================================================
// Kernel 1: persistent LSTM recurrence (512 sequential steps)
//   64 WGs x 256 threads (8 waves). WG g owns hidden units [g*8, g*8+8).
//   W chunk (48KB bf16) is LDS-resident for the whole recurrence; per
//   step, A = [x_t | h_{t-1}] is staged into LDS (h via async LDS loads),
//   then a uniform 12-step bf16-WMMA loop runs entirely out of LDS.
//   waves = (mtile 0..1) x (ntile 0..1) x (khalf 0..1); K-halves reduced
//   with ds_add_f32 into the LDS gate buffer.
// =====================================================================
__global__ void __launch_bounds__(256)
seqlab_lstm_kernel(const int*   __restrict__ q,
                   const int*   __restrict__ lengths,
                   const float* __restrict__ word_embed,
                   const float* __restrict__ lstm_b,
                   const unsigned short* __restrict__ Wt,
                   unsigned short* __restrict__ Hb,      // [2][B_][H_]
                   unsigned short* __restrict__ ys,      // [B_*T_][H_] masked outputs
                   unsigned*       __restrict__ ctr) {
  __shared__ __align__(16) unsigned short Alds[B_ * KIN_];    // [32][768]  48KB
  __shared__ __align__(16) unsigned short Wlds[CPW_ * KIN_];  // [32][768]  48KB
  __shared__ __align__(16) float G[B_][CPW_];                 // gate pre-activations
  __shared__ float Cst[B_][UPW_];                             // cell state (persistent)
  __shared__ float Hst[B_][UPW_];                             // own h (mask copy-through)

  const int tid  = threadIdx.x;
  const int wg   = blockIdx.x;            // 0..63
  const int u0   = wg * UPW_;
  const int lane = tid & 31;
  const int wave = tid >> 5;              // 0..7
  const int mtile = wave & 1;
  const int ntile = (wave >> 1) & 1;
  const int khalf = wave >> 2;
  const int hi    = lane >> 4;            // lane half

  if (tid < B_ * UPW_) { (&Cst[0][0])[tid] = 0.0f; (&Hst[0][0])[tid] = 0.0f; }

  // ---- one-time: load this WG's W chunk into LDS (column-major) ----
  {
    const int c    = tid >> 3;            // local col 0..31
    const int part = tid & 7;
    const int gcol = (c >> 3) * H_ + u0 + (c & 7);     // gate*512 + unit
    const uint4* src = (const uint4*)(Wt + (size_t)gcol * KIN_);
    uint4* dst = (uint4*)(Wlds + c * KIN_);
    #pragma unroll
    for (int i = 0; i < 12; ++i) dst[part * 12 + i] = src[part * 12 + i];
  }

  // per-lane A row (batch) and B column (local gate column)
  const int arow = mtile * 16 + (lane & 15);              // 0..31
  const int cl   = ntile * 16 + (lane & 15);              // local col 0..31

  for (int t = 0; t < T_; ++t) {
    const int rb = t & 1;                 // read buffer (state after step t-1)
    const unsigned short* Hr = Hb + rb * (B_ * H_);
    unsigned short*       Hw = Hb + (rb ^ 1) * (B_ * H_);

    // ---- stage x_t = word_embed[q[:,t]] as bf16 into A cols [0,256) ----
    {
      int b  = tid >> 3;
      int e0 = (tid & 7) * 32;
      const float* src = word_embed + (size_t)q[b * T_ + t] * E_ + e0;
      if (t + 1 < T_)
        __builtin_prefetch(word_embed + (size_t)q[b * T_ + t + 1] * E_ + e0, 0, 1);
      #pragma unroll
      for (int e = 0; e < 32; ++e) Alds[b * KIN_ + e0 + e] = f2bf(src[e]);
    }
    // ---- stage h_{t-1} into A cols [256,768): async global->LDS b128 ----
    #pragma unroll
    for (int i = tid; i < (B_ * H_) / 8; i += 256) {      // 2048 x 16B chunks
      int b = i >> 6, c16 = i & 63;
      copy16_g2l(&Alds[b * KIN_ + E_ + c16 * 8], Hr + b * H_ + c16 * 8);
    }
    // ---- zero gate accumulator ----
    for (int i = tid; i < B_ * CPW_; i += 256) (&G[0][0])[i] = 0.0f;
    wait_async_copies();
    __syncthreads();

    // ---- gate GEMM: uniform 12-step LDS-only WMMA pipeline ----
    v8f acc = {0, 0, 0, 0, 0, 0, 0, 0};
    #pragma unroll
    for (int kk = 0; kk < 12; ++kk) {
      const int koff = khalf * 384 + kk * 32;
      const int o1   = koff + hi * 8;
      Frag a, b;
      a.u[0] = *(const uint4*)&Alds[arow * KIN_ + o1];
      a.u[1] = *(const uint4*)&Alds[arow * KIN_ + o1 + 16];
      const unsigned short* bp = &Wlds[cl * KIN_ + koff + hi * 8];
      b.u[0] = *(const uint4*)bp;
      b.u[1] = *(const uint4*)(bp + 16);
      acc = wmma_bf16(a, b, acc);
    }

    // ---- scatter C fragment into LDS gates (K-half reduction) ----
    #pragma unroll
    for (int k = 0; k < 8; ++k) {
      int brow = mtile * 16 + k + hi * 8;            // C/D layout: M = k + 8*(lane>=16)
      atomicAdd(&G[brow][cl], acc[k]);
    }
    __syncthreads();

    // ---- elementwise LSTM update: one (b, unit) per thread ----
    {
      const int b = tid >> 3;
      const int j = tid & 7;
      const int u = u0 + j;
      float iv = G[b][0 * UPW_ + j] + lstm_b[0 * H_ + u];
      float jv = G[b][1 * UPW_ + j] + lstm_b[1 * H_ + u];
      float fv = G[b][2 * UPW_ + j] + lstm_b[2 * H_ + u] + FORGET_BIAS;
      float ov = G[b][3 * UPW_ + j] + lstm_b[3 * H_ + u];
      float c_old = Cst[b][j], h_old = Hst[b][j];
      float c_new = c_old * sigm(fv) + sigm(iv) * tanhf(jv);
      float h_new = tanhf(c_new) * sigm(ov);
      float m  = (t < lengths[b]) ? 1.0f : 0.0f;
      float c2 = m * c_new + (1.0f - m) * c_old;
      float h2 = m * h_new + (1.0f - m) * h_old;
      Cst[b][j] = c2;
      Hst[b][j] = h2;
      Hw[b * H_ + u] = f2bf(h2);
      ys[((size_t)b * T_ + t) * H_ + u] = f2bf(m * h_new);
    }
    grid_barrier(ctr, (unsigned)(t + 1) * NWG_);
  }
}

// =====================================================================
// Kernel 2: projection + relu + log-softmax + masked xent per row
//   512 WGs x 256 threads; each WG handles 32 rows x 64 (padded) cols.
// =====================================================================
__global__ void __launch_bounds__(256)
seqlab_proj_kernel(const int* __restrict__ alab,
                   const int* __restrict__ lengths,
                   const unsigned short* __restrict__ ys,  // [B_*T_][H_] bf16
                   const unsigned short* __restrict__ Pt,  // [LPAD_][H_] bf16
                   const float* __restrict__ proj_b,
                   float* __restrict__ xent) {
  __shared__ __align__(16) float Llds[32][LPAD_];

  const int tid  = threadIdx.x;
  const int lane = tid & 31;
  const int wave = tid >> 5;            // 0..7
  const int mtile = wave & 1;
  const int nt    = wave >> 1;          // 0..3
  const int hi    = lane >> 4;
  const int rowbase = blockIdx.x * 32;

  const int arow = rowbase + mtile * 16 + (lane & 15);
  const int col  = nt * 16 + (lane & 15);
  const unsigned short* pcol = Pt + (size_t)col * H_;

  v8f acc = {0, 0, 0, 0, 0, 0, 0, 0};
  #pragma unroll
  for (int kk = 0; kk < 16; ++kk) {
    const int koff = kk * 32;
    const int o1   = koff + hi * 8;
    Frag a, b;
    const unsigned short* ap = ys + (size_t)arow * H_ + o1;
    a.u[0] = *(const uint4*)ap;
    a.u[1] = *(const uint4*)(ap + 16);
    const unsigned short* bp = pcol + koff + hi * 8;
    b.u[0] = *(const uint4*)bp;
    b.u[1] = *(const uint4*)(bp + 16);
    acc = wmma_bf16(a, b, acc);
  }
  #pragma unroll
  for (int k = 0; k < 8; ++k) {
    int brow = mtile * 16 + k + hi * 8;
    Llds[brow][col] = acc[k];
  }
  __syncthreads();

  if (tid < 32) {
    const int row = rowbase + tid;
    const int b = row / T_, t = row % T_;
    float mx = -1.0e30f;
    #pragma unroll 1
    for (int c = 0; c < L_; ++c) {
      float v = fmaxf(Llds[tid][c] + proj_b[c], 0.0f);   // relu(logit)
      mx = fmaxf(mx, v);
    }
    float s = 0.0f;
    #pragma unroll 1
    for (int c = 0; c < L_; ++c) {
      float v = fmaxf(Llds[tid][c] + proj_b[c], 0.0f);
      s += __expf(v - mx);
    }
    const int lab = alab[row];
    float vl = fmaxf(Llds[tid][lab] + proj_b[lab], 0.0f);
    float xe = (__logf(s) + mx) - vl;                    // -log_softmax[label]
    float m  = (t < lengths[b]) ? 1.0f : 0.0f;
    xent[row] = xe * m;
  }
}

// =====================================================================
// Kernel 3: final reduction  loss = (1/B) * sum_b (sum_t xent[b,t]) / len[b]
// =====================================================================
__global__ void seqlab_loss_kernel(const int* __restrict__ lengths,
                                   const float* __restrict__ xent,
                                   float* __restrict__ out) {
  __shared__ float red[256];
  const int tid = threadIdx.x;
  float acc = 0.0f;
  for (int i = tid; i < B_ * T_; i += 256) {
    int b = i >> 9;                       // i / T_
    acc += xent[i] / (float)lengths[b];
  }
  red[tid] = acc;
  __syncthreads();
  for (int s = 128; s > 0; s >>= 1) {
    if (tid < s) red[tid] += red[tid + s];
    __syncthreads();
  }
  if (tid == 0) out[0] = red[0] / (float)B_;
}

// =====================================================================
// Host-side launch
// =====================================================================
extern "C" void kernel_launch(void* const* d_in, const int* in_sizes, int n_in,
                              void* d_out, int out_size, void* d_ws, size_t ws_size,
                              hipStream_t stream) {
  (void)in_sizes; (void)n_in; (void)out_size; (void)ws_size;
  const int*   q          = (const int*)  d_in[0];
  const int*   a          = (const int*)  d_in[1];
  const int*   lengths    = (const int*)  d_in[2];
  const float* word_embed = (const float*)d_in[3];
  const float* lstm_W     = (const float*)d_in[4];
  const float* lstm_b     = (const float*)d_in[5];
  const float* proj_W     = (const float*)d_in[6];
  const float* proj_b     = (const float*)d_in[7];
  float* out = (float*)d_out;

  // workspace carve-out (~20.3 MB total)
  char* ws = (char*)d_ws;
  size_t off = 0;
  auto carve = [&](size_t bytes) -> char* {
    char* p = ws + off;
    off = (off + bytes + 255) & ~(size_t)255;
    return p;
  };
  unsigned short* Wt   = (unsigned short*)carve((size_t)N4H_ * KIN_ * 2);
  unsigned short* Pt   = (unsigned short*)carve((size_t)LPAD_ * H_ * 2);
  unsigned short* Hb   = (unsigned short*)carve((size_t)2 * B_ * H_ * 2);
  unsigned short* ysb  = (unsigned short*)carve((size_t)B_ * T_ * H_ * 2);
  float*          xent = (float*)         carve((size_t)B_ * T_ * 4);
  unsigned*       ctr  = (unsigned*)      carve(256);

  const int initN = (N4H_ * KIN_ + 255) / 256;
  seqlab_init_kernel<<<initN, 256, 0, stream>>>(lstm_W, proj_W, Wt, Pt, Hb, ctr);
  seqlab_lstm_kernel<<<NWG_, 256, 0, stream>>>(q, lengths, word_embed, lstm_b,
                                               Wt, Hb, ysb, ctr);
  seqlab_proj_kernel<<<(B_ * T_) / 32, 256, 0, stream>>>(a, lengths, ysb, Pt,
                                                         proj_b, xent);
  seqlab_loss_kernel<<<1, 256, 0, stream>>>(lengths, xent, out);
}
